// Attention_44848048504871
// MI455X (gfx1250) — matrix operand order repaired and checked by reading, not compiled, not run on hardware
//
#include <hip/hip_runtime.h>
#include <hip/hip_bf16.h>
#include <math.h>

typedef _Float16 v16h __attribute__((ext_vector_type(16)));
typedef _Float16 v8h  __attribute__((ext_vector_type(8)));
typedef _Float16 v4h  __attribute__((ext_vector_type(4)));
typedef float    v8f  __attribute__((ext_vector_type(8)));
typedef float    v4f  __attribute__((ext_vector_type(4)));

#define BB 32
#define LL 2048
#define HH 1024
#define MT 64            // key rows per block
#define LDK 1032         // padded LDS row stride in halves (2064 B, 16B aligned)

// gfx1250 has V_TANH_F32 (TRANS32); use the builtin if this clang exposes it.
#if __has_builtin(__builtin_amdgcn_tanhf)
#define DEV_TANH(x) __builtin_amdgcn_tanhf(x)
#else
#define DEV_TANH(x) tanhf(x)
#endif

// ---------------- Kernel 1: qpb[b,o] = sum_h query[b,h]*Wa_w[o,h] + Wa_b[o] + Ua_b[o]
__global__ void qproj_kernel(const float* __restrict__ query,
                             const float* __restrict__ Wa_w,
                             const float* __restrict__ Wa_b,
                             const float* __restrict__ Ua_b,
                             float* __restrict__ qpb) {
    __shared__ float qs[HH];
    const int b = blockIdx.x;
    const int tid = threadIdx.x;   // 256 threads
    for (int i = tid; i < HH; i += 256) qs[i] = query[b * HH + i];
    __syncthreads();
    #pragma unroll
    for (int j = 0; j < 4; ++j) {
        const int o = tid + j * 256;
        const float* wr = Wa_w + (size_t)o * HH;
        float acc = 0.f;
        #pragma unroll 4
        for (int h = 0; h < HH; ++h) acc += qs[h] * wr[h];
        qpb[b * HH + o] = acc + Wa_b[o] + Ua_b[o];
    }
}

// ---------------- Kernel 2: fp32 -> f16 conversion (Ua_w)
__global__ void cvt_kernel(const float* __restrict__ src, _Float16* __restrict__ dst) {
    const int i = (blockIdx.x * blockDim.x + threadIdx.x) * 4;
    v4f v = *(const v4f*)(src + i);
    v4h h = { (_Float16)v.x, (_Float16)v.y, (_Float16)v.z, (_Float16)v.w };
    *(v4h*)(dst + i) = h;
}

// ---------------- Kernel 3: fused k-projection (WMMA f16) + tanh + va reduction -> score[b,l]
__global__ void __launch_bounds__(256)
score_kernel(const float* __restrict__ key,
             const _Float16* __restrict__ ua16,
             const float* __restrict__ qpb,
             const float* __restrict__ va_w,
             const float* __restrict__ va_b,
             float* __restrict__ score) {
    extern __shared__ _Float16 smem[];          // keyLds[MT][LDK] f16
    __shared__ float scoreLds[2][MT];

    const int tid  = threadIdx.x;
    const int wave = tid >> 5;                  // 0..7
    const int lane = tid & 31;
    const int rt   = wave & 3;                  // row-tile 0..3
    const int ct   = wave >> 2;                 // col-tile 0..1
    const int n    = lane & 15;                 // A row-in-tile m AND B column n
    const int hi   = lane >> 4;

    const int m0 = blockIdx.x * MT;             // global row (b*L + l)
    const int b  = m0 >> 11;                    // / 2048
    const int l0 = m0 & (LL - 1);

    // ---- stage 64 key rows into LDS as f16 (256 floats per thread, float4 loads)
    {
        const float* kbase = key + (size_t)m0 * HH;
        #pragma unroll 4
        for (int it = 0; it < (MT * HH) / (4 * 256); ++it) {   // 64 iters
            const int idx = it * 256 + tid;                    // float4 index
            const int row = idx >> 8;                          // 256 float4 per row
            const int kk  = (idx & 255) * 4;
            v4f v = *(const v4f*)(kbase + (size_t)row * HH + kk);
            v4h h = { (_Float16)v.x, (_Float16)v.y, (_Float16)v.z, (_Float16)v.w };
            *(v4h*)(smem + row * LDK + kk) = h;
        }
    }

    // ---- preload this lane's (qpb + biases) and va values for all 32 o-steps
    float qv[32], vv[32];
    {
        const int oc0 = ct * 16 + n;
        #pragma unroll
        for (int i = 0; i < 32; ++i) {
            qv[i] = qpb[b * HH + i * 32 + oc0];
            vv[i] = va_w[i * 32 + oc0];
        }
    }
    __syncthreads();

    float sacc[8];
    #pragma unroll
    for (int r = 0; r < 8; ++r) sacc[r] = 0.f;

    // A operand: this lane holds key row (rt*16 + n), K chunks per ISA layout
    const _Float16* aBase = smem + (rt * 16 + n) * LDK;

    for (int os = 0; os < 32; ++os) {
        const int ocol = os * 32 + ct * 16 + n;                // Ua row == output column
        const _Float16* bBase = ua16 + (size_t)ocol * HH;
        v8f c = {};
        #pragma unroll 8
        for (int kk = 0; kk < HH; kk += 32) {
            // A: lane m, halves 0..7 = K(kk + hi*8 ..), halves 8..15 = K(kk+16+hi*8 ..)
            v8h a0 = *(const v8h*)(aBase + kk + hi * 8);
            v8h a1 = *(const v8h*)(aBase + kk + 16 + hi * 8);
            v16h A = __builtin_shufflevector(a0, a1, 0,1,2,3,4,5,6,7,8,9,10,11,12,13,14,15);
            // B: lane n = column n, halves 0..7 = K(kk + hi*8 ..), halves 8..15 = K(kk+16+hi*8 ..)
            v8h b0 = *(const v8h*)(bBase + kk + hi * 8);
            v8h b1 = *(const v8h*)(bBase + kk + 16 + hi * 8);
            v16h Bm = __builtin_shufflevector(b0, b1, 0,1,2,3,4,5,6,7,8,9,10,11,12,13,14,15);
            c = __builtin_amdgcn_wmma_f32_16x16x32_f16(false, A, false, Bm,
                                                       (short)0, c, false, false);
        }
        // fuse: sacc[r] += va[ocol] * tanh(qpb[b,ocol] + kproj)   (C: row r+8*hi, col n)
        const float q = qv[os];
        const float w = vv[os];
        #pragma unroll
        for (int r = 0; r < 8; ++r)
            sacc[r] += w * DEV_TANH(q + c[r]);
    }

    // reduce across the 16 columns (lanes n=0..15 within each half-wave; hi preserved)
    #pragma unroll
    for (int r = 0; r < 8; ++r) {
        float v = sacc[r];
        v += __shfl_xor(v, 1);
        v += __shfl_xor(v, 2);
        v += __shfl_xor(v, 4);
        v += __shfl_xor(v, 8);
        sacc[r] = v;
    }
    if (n == 0) {
        #pragma unroll
        for (int r = 0; r < 8; ++r)
            scoreLds[ct][rt * 16 + r + 8 * hi] = sacc[r];
    }
    __syncthreads();
    if (tid < MT)
        score[(size_t)b * LL + l0 + tid] = scoreLds[0][tid] + scoreLds[1][tid] + va_b[0];
}

// ---------------- Kernel 4: masked softmax over L per batch -> attn weights
__global__ void softmax_kernel(const float* __restrict__ score,
                               const int* __restrict__ mask,
                               float* __restrict__ attn) {
    __shared__ float red[256];
    const int b = blockIdx.x, tid = threadIdx.x;
    float s[8];
    float mx = -3.4e38f;
    #pragma unroll
    for (int j = 0; j < 8; ++j) {
        const int l = j * 256 + tid;
        float v = score[b * LL + l];
        s[j] = (mask[b * LL + l] == 0) ? -1e10f : v;
        mx = fmaxf(mx, s[j]);
    }
    red[tid] = mx; __syncthreads();
    for (int st = 128; st > 0; st >>= 1) {
        if (tid < st) red[tid] = fmaxf(red[tid], red[tid + st]);
        __syncthreads();
    }
    const float MX = red[0];
    __syncthreads();
    float lsum = 0.f;
    #pragma unroll
    for (int j = 0; j < 8; ++j) { s[j] = expf(s[j] - MX); lsum += s[j]; }
    red[tid] = lsum; __syncthreads();
    for (int st = 128; st > 0; st >>= 1) {
        if (tid < st) red[tid] += red[tid + st];
        __syncthreads();
    }
    const float inv = 1.f / red[0];
    #pragma unroll
    for (int j = 0; j < 8; ++j)
        attn[b * LL + j * 256 + tid] = s[j] * inv;
}

// ---------------- Kernel 5: context[b,h] = sum_l attn[b,l] * value[b,l,h]
__global__ void context_kernel(const float* __restrict__ attn,
                               const float* __restrict__ value,
                               float* __restrict__ out) {
    const int b = blockIdx.x >> 2;
    const int h = (blockIdx.x & 3) * 256 + threadIdx.x;
    const float* vb = value + (size_t)b * LL * HH + h;
    const float* ab = attn + b * LL;
    float acc = 0.f;
    #pragma unroll 8
    for (int l = 0; l < LL; ++l) acc += ab[l] * vb[(size_t)l * HH];
    out[b * HH + h] = acc;
}

extern "C" void kernel_launch(void* const* d_in, const int* in_sizes, int n_in,
                              void* d_out, int out_size, void* d_ws, size_t ws_size,
                              hipStream_t stream) {
    const float* query = (const float*)d_in[0];
    const float* key   = (const float*)d_in[1];
    const float* value = (const float*)d_in[2];
    const int*   mask  = (const int*)d_in[3];
    const float* Wa_w  = (const float*)d_in[4];
    const float* Wa_b  = (const float*)d_in[5];
    const float* Ua_w  = (const float*)d_in[6];
    const float* Ua_b  = (const float*)d_in[7];
    const float* va_w  = (const float*)d_in[8];
    const float* va_b  = (const float*)d_in[9];
    float* out = (float*)d_out;

    char* ws = (char*)d_ws;
    float*    qpb   = (float*)ws;                                   // 32*1024 f32   (128 KB)
    _Float16* ua16  = (_Float16*)(ws + (32 * 1024) * 4);            // 1024*1024 f16 (2 MB)
    float*    score = (float*)(ws + (32 * 1024) * 4 + (1024 * 1024) * 2);       // 32*2048 f32
    float*    attn  = (float*)(ws + (32 * 1024) * 4 + (1024 * 1024) * 2 + (32 * 2048) * 4);

    qproj_kernel<<<BB, 256, 0, stream>>>(query, Wa_w, Wa_b, Ua_b, qpb);
    cvt_kernel<<<(HH * HH) / (4 * 256), 256, 0, stream>>>(Ua_w, ua16);
    const size_t smemBytes = (size_t)MT * LDK * sizeof(_Float16);   // ~129 KB dynamic LDS
    score_kernel<<<(BB * LL) / MT, 256, smemBytes, stream>>>(key, ua16, qpb, va_w, va_b, score);
    softmax_kernel<<<BB, 256, 0, stream>>>(score, mask, attn);
    context_kernel<<<BB * (HH / 256), 256, 0, stream>>>(attn, value, out);
}
